// Attention_3143916061330
// MI455X (gfx1250) — matrix operand order repaired and checked by reading, not compiled, not run on hardware
//
#include <hip/hip_runtime.h>

#define HEADS 16
#define HSIZE 64
#define DIMN  1024
#define SEQ   1024
#define BATCH 4
#define NTOK  (BATCH * SEQ) /* 4096 */
#define NINFV -10000.0f

typedef __attribute__((ext_vector_type(16))) _Float16 v16h;
typedef __attribute__((ext_vector_type(8)))  _Float16 v8h;
typedef __attribute__((ext_vector_type(8)))  float    v8f;
typedef __attribute__((ext_vector_type(4)))  int      v4i;

#define DEV static __device__ __forceinline__

// ---------------- async global->LDS staging (CDNA5 ASYNCcnt path) ----------
#if __has_builtin(__builtin_amdgcn_global_load_async_to_lds_b128)
#define HAVE_ASYNC_LDS 1
#else
#define HAVE_ASYNC_LDS 0
#endif

typedef __attribute__((address_space(1))) v4i g_v4i;  // global int4
typedef __attribute__((address_space(3))) v4i l_v4i;  // LDS int4

DEV void cp16(_Float16* lds, const _Float16* g) {
#if HAVE_ASYNC_LDS
  __builtin_amdgcn_global_load_async_to_lds_b128((g_v4i*)g, (l_v4i*)lds, 0, 0);
#else
  *(v8h*)lds = *(const v8h*)g;
#endif
}

DEV void cp_drain() {
#if HAVE_ASYNC_LDS
#if __has_builtin(__builtin_amdgcn_s_wait_asynccnt)
  __builtin_amdgcn_s_wait_asynccnt(0);
#else
  asm volatile("s_wait_asynccnt 0x0" ::: "memory");
#endif
#endif
}

// ----------------------------------------------------------- fragments -----
DEV v16h pack16(v8h lo, v8h hi) {
  v16h r;
#pragma unroll
  for (int i = 0; i < 8; ++i) { r[i] = lo[i]; r[i + 8] = hi[i]; }
  return r;
}

// A-fragment (16x32 f16, MxK) from row-major storage, leading dim ldh halves.
DEV v16h frag_a(const _Float16* base, int ldh) {
  int lane = threadIdx.x & 31;
  const _Float16* p = base + (lane & 15) * ldh + (lane >> 4) * 8;
  return pack16(*(const v8h*)p, *(const v8h*)(p + 16));
}

// B-fragment (32x16 f16, KxN) from N-major storage bt[n][k], leading dim ldh.
DEV v16h frag_b(const _Float16* bt, int ldh) {
  int lane = threadIdx.x & 31;
  const _Float16* p = bt + (lane & 15) * ldh + (lane >> 4) * 8;
  return pack16(*(const v8h*)p, *(const v8h*)(p + 16));
}

DEV v8f wmma_f16(v16h a, v16h b, v8f c) {
  return __builtin_amdgcn_wmma_f32_16x16x32_f16(false, a, false, b, (short)0, c,
                                                false, false);
}

// ---------------------------------------------------------------- casts -----
__global__ void cast_f32_f16(const float* __restrict__ s,
                             _Float16* __restrict__ d, int n) {
  int i = blockIdx.x * blockDim.x + threadIdx.x;
  if (i < n) d[i] = (_Float16)s[i];
}

// Tiled transpose + cast: dst[n][k] = (f16)src[k][n], 1024x1024.
__global__ void transpose_cast(const float* __restrict__ src,
                               _Float16* __restrict__ dst) {
  __shared__ float tile[32][33];
  int nx = blockIdx.x * 32 + threadIdx.x;  // n (read coalesced)
  int k0 = blockIdx.y * 32;
#pragma unroll
  for (int i = 0; i < 32; i += 8)
    tile[threadIdx.y + i][threadIdx.x] =
        src[(size_t)(k0 + threadIdx.y + i) * DIMN + nx];
  __syncthreads();
  int ko = k0 + threadIdx.x;               // k (write coalesced)
  int n0 = blockIdx.x * 32;
#pragma unroll
  for (int i = 0; i < 32; i += 8)
    dst[(size_t)(n0 + threadIdx.y + i) * DIMN + ko] =
        (_Float16)tile[threadIdx.x][threadIdx.y + i];
}

// ---------------------------------------------------------------- GEMM ------
// C = A(f16 [NTOK][DIMN]) * B, where Wt is PRE-TRANSPOSED: Wt[n][k] (f16).
// MODE 0: f16 head-major  out[((b*16+h)*SEQ+s)*64+dh]            (Q, K)
// MODE 2: f16 head-major TRANSPOSED out[((b*16+h)*64+dh)*SEQ+s]  (V)
// MODE 1: f32 outR[m][n] = C + bias + x residual                  (O)
template <int MODE>
__global__ void gemm_kernel(const _Float16* __restrict__ A,
                            const _Float16* __restrict__ Wt,
                            const float* __restrict__ bias,
                            const float* __restrict__ xres,
                            _Float16* __restrict__ outH,
                            float* __restrict__ outR) {
  __shared__ _Float16 sA[128 * 72];  // 128 rows x 64 k, ld 72
  __shared__ _Float16 sB[64 * 72];   // 64 n-rows x 64 k, ld 72
  const int tid  = threadIdx.x;
  const int wave = tid >> 5, lane = tid & 31;
  const int half = lane >> 4, n16 = lane & 15;
  const int mb = blockIdx.y * 128, nb = blockIdx.x * 64;

  v8f acc[4];
#pragma unroll
  for (int j = 0; j < 4; ++j)
#pragma unroll
    for (int r = 0; r < 8; ++r) acc[j][r] = 0.0f;

  for (int kb = 0; kb < DIMN; kb += 64) {
    // stage A tile 128x64 as 16B async chunks (4 per thread)
#pragma unroll
    for (int i = 0; i < 4; ++i) {
      int li = i * 256 + tid;  // 1024 chunks
      int row = li >> 3, c8 = (li & 7) * 8;
      cp16(sA + row * 72 + c8, A + (size_t)(mb + row) * DIMN + kb + c8);
    }
    // stage B tile (n-major) 64x64 as 16B async chunks (2 per thread)
#pragma unroll
    for (int i = 0; i < 2; ++i) {
      int li = i * 256 + tid;  // 512 chunks
      int row = li >> 3, c8 = (li & 7) * 8;
      cp16(sB + row * 72 + c8, Wt + (size_t)(nb + row) * DIMN + kb + c8);
    }
    cp_drain();
    __syncthreads();
#pragma unroll
    for (int ks = 0; ks < 2; ++ks) {
      v16h af = frag_a(sA + wave * 16 * 72 + ks * 32, 72);
#pragma unroll
      for (int j = 0; j < 4; ++j) {
        v16h bf = frag_b(sB + j * 16 * 72 + ks * 32, 72);
        acc[j] = wmma_f16(af, bf, acc[j]);
      }
    }
    __syncthreads();
  }

#pragma unroll
  for (int j = 0; j < 4; ++j) {
    int n = nb + j * 16 + n16;
    float bv = bias[n];
#pragma unroll
    for (int r = 0; r < 8; ++r) {
      int m = mb + wave * 16 + r + 8 * half;
      float v = acc[j][r] + bv;
      int b = m >> 10, s = m & 1023, h = n >> 6, dh = n & 63;
      if (MODE == 0) {
        outH[(((size_t)(b * HEADS + h)) * SEQ + s) * HSIZE + dh] = (_Float16)v;
      } else if (MODE == 2) {
        outH[(((size_t)(b * HEADS + h)) * HSIZE + dh) * SEQ + s] = (_Float16)v;
      } else {
        outR[(size_t)m * DIMN + n] = v + xres[(size_t)m * DIMN + n];
      }
    }
  }
}

// ------------------------------------------------------------- attention ----
#define K_LD 72
#define V_LD 1032
#define P_LD 40
#define OFF_V (SEQ * K_LD)                 /* 73728 halves */
#define OFF_P (OFF_V + HSIZE * V_LD)       /* 139776 halves */
#define SMEM_HALVES (OFF_P + 8 * 16 * P_LD)
#define SMEM_BYTES (SMEM_HALVES * 2)       /* 289792 B < 320 KB WGP LDS */

__global__ void attn_kernel(const _Float16* __restrict__ qh,
                            const _Float16* __restrict__ kh,
                            const _Float16* __restrict__ vt,
                            const float* __restrict__ rel,
                            const int* __restrict__ mask,
                            float* __restrict__ attn,
                            _Float16* __restrict__ ctxh) {
  extern __shared__ _Float16 smem[];
  _Float16* sK = smem;           // [SEQ][K_LD]   K row-major
  _Float16* sV = smem + OFF_V;   // [HSIZE][V_LD] V already transposed
  _Float16* sP = smem + OFF_P;   // per-wave 16x32 prob bounce tiles

  const int tid  = threadIdx.x;
  const int wave = tid >> 5, lane = tid & 31;
  const int half = lane >> 4, n16 = lane & 15;
  const int mblk = blockIdx.x, bh = blockIdx.y;
  const int b = bh >> 4, h = bh & 15;

  const _Float16* kb = kh + (size_t)bh * SEQ * HSIZE;   // [t][d]
  const _Float16* vb = vt + (size_t)bh * HSIZE * SEQ;   // [dh][t]

  // stage whole K and V^T via async global->LDS (16B chunks, with pad)
  for (int c = tid; c < SEQ * 8; c += 256) {
    int row = c >> 3, col = (c & 7) * 8;
    cp16(sK + row * K_LD + col, kb + row * HSIZE + col);
  }
  for (int c = tid; c < HSIZE * 128; c += 256) {
    int row = c >> 7, col = (c & 127) * 8;
    cp16(sV + row * V_LD + col, vb + row * SEQ + col);
  }
  cp_drain();
  __syncthreads();

  const int s0 = mblk * 128 + wave * 16;
  const _Float16* qbase = qh + ((size_t)bh * SEQ + s0) * HSIZE;
  v16h aQ0 = frag_a(qbase, HSIZE);       // q d=0..31
  v16h aQ1 = frag_a(qbase + 32, HSIZE);  // q d=32..63

  const float* relb = rel + ((size_t)h * SEQ + s0) * SEQ;
  const int* maskb = mask + b * SEQ;

  auto score = [&](int t0) -> v8f {
    v8f c;
#pragma unroll
    for (int r = 0; r < 8; ++r) c[r] = 0.0f;
    c = wmma_f16(aQ0, frag_b(sK + t0 * K_LD, K_LD), c);
    c = wmma_f16(aQ1, frag_b(sK + t0 * K_LD + 32, K_LD), c);
    int t = t0 + n16;
    float mk = maskb[t] ? NINFV : 0.0f;
#pragma unroll
    for (int r = 0; r < 8; ++r) {
      int row = r + 8 * half;
      c[r] = c[r] * 0.125f + relb[(size_t)row * SEQ + t] + mk;
    }
    return c;
  };

  // ---- pass 1: LANE-LOCAL online max/sum (no per-tile cross-lane traffic) --
  float m_run[8], l_run[8];
#pragma unroll
  for (int r = 0; r < 8; ++r) { m_run[r] = -3.0e38f; l_run[r] = 0.0f; }

  for (int t0 = 0; t0 < SEQ; t0 += 16) {
    v8f s = score(t0);
#pragma unroll
    for (int r = 0; r < 8; ++r) {
      float mo = m_run[r];
      float mn = fmaxf(mo, s[r]);
      l_run[r] = l_run[r] * __expf(mo - mn) + __expf(s[r] - mn);
      m_run[r] = mn;
    }
  }

  // single cross-lane reduction over the 16 lanes of each half-wave
  float rinv[8];
#pragma unroll
  for (int r = 0; r < 8; ++r) {
    float M = m_run[r];
    M = fmaxf(M, __shfl_xor(M, 1));
    M = fmaxf(M, __shfl_xor(M, 2));
    M = fmaxf(M, __shfl_xor(M, 4));
    M = fmaxf(M, __shfl_xor(M, 8));
    float L = l_run[r] * __expf(m_run[r] - M);
    L += __shfl_xor(L, 1);
    L += __shfl_xor(L, 2);
    L += __shfl_xor(L, 4);
    L += __shfl_xor(L, 8);
    m_run[r] = M;
    rinv[r] = 1.0f / L;
  }

  // ---- pass 2: write attn, accumulate ctx = P @ V ----
  v8f cctx[4];
#pragma unroll
  for (int j = 0; j < 4; ++j)
#pragma unroll
    for (int r = 0; r < 8; ++r) cctx[j][r] = 0.0f;

  _Float16* myP = sP + wave * 16 * P_LD;  // wave-private bounce tile
  float* attnb = attn + ((size_t)bh * SEQ + s0) * SEQ;

  for (int tt = 0; tt < SEQ; tt += 32) {
#pragma unroll
    for (int par = 0; par < 2; ++par) {
      int t0 = tt + par * 16;
      v8f s = score(t0);
#pragma unroll
      for (int r = 0; r < 8; ++r) {
        float p = __expf(s[r] - m_run[r]) * rinv[r];
        int row = r + 8 * half;
        attnb[(size_t)row * SEQ + t0 + n16] = p;
        myP[row * P_LD + par * 16 + n16] = (_Float16)p;  // C-layout -> LDS
      }
    }
    v16h aP = frag_a(myP, P_LD);  // LDS -> A-layout
#pragma unroll
    for (int j = 0; j < 4; ++j) {
      v16h bf = frag_b(sV + j * 16 * V_LD + tt, V_LD);
      cctx[j] = wmma_f16(aP, bf, cctx[j]);
    }
  }

#pragma unroll
  for (int j = 0; j < 4; ++j)
#pragma unroll
    for (int r = 0; r < 8; ++r) {
      int row = s0 + r + 8 * half;
      int dh = j * 16 + n16;
      ctxh[((size_t)(b * SEQ + row)) * DIMN + h * HSIZE + dh] =
          (_Float16)cctx[j][r];
    }
}

// --------------------------------------------------------------- layernorm --
__global__ void ln_kernel(float* __restrict__ io,
                          const float* __restrict__ gamma,
                          const float* __restrict__ beta) {
  __shared__ float red[256];
  int row = blockIdx.x, tid = threadIdx.x;
  float* p = io + (size_t)row * DIMN;
  float v[4];
#pragma unroll
  for (int i = 0; i < 4; ++i) v[i] = p[tid + i * 256];
  red[tid] = v[0] + v[1] + v[2] + v[3];
  __syncthreads();
  for (int o = 128; o > 0; o >>= 1) {
    if (tid < o) red[tid] += red[tid + o];
    __syncthreads();
  }
  float mu = red[0] * (1.0f / DIMN);
  __syncthreads();
  float q = 0.0f;
#pragma unroll
  for (int i = 0; i < 4; ++i) { float d = v[i] - mu; q += d * d; }
  red[tid] = q;
  __syncthreads();
  for (int o = 128; o > 0; o >>= 1) {
    if (tid < o) red[tid] += red[tid + o];
    __syncthreads();
  }
  float rstd = rsqrtf(red[0] * (1.0f / DIMN) + 1e-6f);
#pragma unroll
  for (int i = 0; i < 4; ++i) {
    int c = tid + i * 256;
    p[c] = (v[i] - mu) * rstd * gamma[c] + beta[c];
  }
}

// ----------------------------------------------------------------- launch ---
extern "C" void kernel_launch(void* const* d_in, const int* in_sizes, int n_in,
                              void* d_out, int out_size, void* d_ws,
                              size_t ws_size, hipStream_t stream) {
  (void)in_sizes; (void)n_in; (void)out_size; (void)ws_size;
  const float* x     = (const float*)d_in[0];
  const int*   mask  = (const int*)d_in[1];
  const float* rel   = (const float*)d_in[2];
  const float* Wq    = (const float*)d_in[3];
  const float* bq    = (const float*)d_in[4];
  const float* Wk    = (const float*)d_in[5];
  const float* bk    = (const float*)d_in[6];
  const float* Wv    = (const float*)d_in[7];
  const float* bv    = (const float*)d_in[8];
  const float* Wo    = (const float*)d_in[9];
  const float* bo    = (const float*)d_in[10];
  const float* gamma = (const float*)d_in[11];
  const float* beta  = (const float*)d_in[12];

  float* out  = (float*)d_out;              // [4096][1024] f32 (res scratch)
  float* attn = out + (size_t)NTOK * DIMN;  // [64][1024][1024] f32

  char* ws = (char*)d_ws;                          // 48 MB used
  _Float16* xh   = (_Float16*)(ws);                //  8 MB
  _Float16* wqt  = (_Float16*)(ws + (8u  << 20));  //  2 MB each (transposed)
  _Float16* wkt  = (_Float16*)(ws + (10u << 20));
  _Float16* wvt  = (_Float16*)(ws + (12u << 20));
  _Float16* wot  = (_Float16*)(ws + (14u << 20));
  _Float16* qhp  = (_Float16*)(ws + (16u << 20));  //  8 MB, [bh][s][dh]
  _Float16* khp  = (_Float16*)(ws + (24u << 20));  //  8 MB, [bh][t][dh]
  _Float16* vtp  = (_Float16*)(ws + (32u << 20));  //  8 MB, [bh][dh][t]
  _Float16* ctxh = (_Float16*)(ws + (40u << 20));  //  8 MB

  cast_f32_f16<<<NTOK * DIMN / 256, 256, 0, stream>>>(x, xh, NTOK * DIMN);
  dim3 tgrid(DIMN / 32, DIMN / 32), tblk(32, 8);
  transpose_cast<<<tgrid, tblk, 0, stream>>>(Wq, wqt);
  transpose_cast<<<tgrid, tblk, 0, stream>>>(Wk, wkt);
  transpose_cast<<<tgrid, tblk, 0, stream>>>(Wv, wvt);
  transpose_cast<<<tgrid, tblk, 0, stream>>>(Wo, wot);

  dim3 ggrid(DIMN / 64, NTOK / 128);  // (16, 32)
  gemm_kernel<0><<<ggrid, 256, 0, stream>>>(xh, wqt, bq, nullptr, qhp, nullptr);
  gemm_kernel<0><<<ggrid, 256, 0, stream>>>(xh, wkt, bk, nullptr, khp, nullptr);
  gemm_kernel<2><<<ggrid, 256, 0, stream>>>(xh, wvt, bv, nullptr, vtp, nullptr);

  static_assert(SMEM_BYTES <= 320 * 1024, "LDS budget");
  (void)hipFuncSetAttribute(reinterpret_cast<const void*>(attn_kernel),
                            hipFuncAttributeMaxDynamicSharedMemorySize,
                            SMEM_BYTES);
  attn_kernel<<<dim3(SEQ / 128, BATCH * HEADS), 256, SMEM_BYTES, stream>>>(
      qhp, khp, vtp, rel, mask, attn, ctxh);

  gemm_kernel<1><<<ggrid, 256, 0, stream>>>(ctxh, wot, bo, x, nullptr, out);
  ln_kernel<<<NTOK, 256, 0, stream>>>(out, gamma, beta);
}
